// MultiHeadWassersteinAttention_21165598835372
// MI455X (gfx1250) — matrix operand order repaired and checked by reading, not compiled, not run on hardware
//
#include <hip/hip_runtime.h>

typedef __attribute__((ext_vector_type(16))) __bf16 v16bf;
typedef __attribute__((ext_vector_type(8))) float v8f;

union FragU { v16bf v; uint4 u[2]; };

__device__ inline unsigned short f32_to_bf16(float f) {
  unsigned int u = __float_as_uint(f);
  unsigned int r = u + 0x7FFFu + ((u >> 16) & 1u);
  return (unsigned short)(r >> 16);
}

__device__ inline float bf16_to_f32(unsigned short h) {
  return __uint_as_float(((unsigned int)h) << 16);
}

__device__ inline v8f zero_v8f() {
  v8f z;
#pragma unroll
  for (int i = 0; i < 8; ++i) z[i] = 0.f;
  return z;
}

// A fragment: 16x32 bf16, row-major tile in LDS (stride in elements).
// Lane half h: elements 0..7 -> K = 8h..8h+7 ; elements 8..15 -> K = 16+8h..16+8h+7.
__device__ inline v16bf load_a_frag(const unsigned short* smem, int row0, int k0, int stride) {
  const int lane = threadIdx.x & 31;
  const int hl = lane >> 4;
  const int m = lane & 15;
  const unsigned short* p = smem + (row0 + m) * stride + k0;
  FragU f;
  f.u[0] = *(const uint4*)(p + 8 * hl);
  f.u[1] = *(const uint4*)(p + 16 + 8 * hl);
  return f.v;
}

// B fragment from an LDS tile stored [n][k] (K contiguous per column).
// Lane half h: elements 0..7 -> K = 8h..8h+7 ; elements 8..15 -> K = 16+8h..16+8h+7, column = lane&15.
__device__ inline v16bf load_bt_frag(const unsigned short* smem, int col0, int k0, int stride) {
  const int lane = threadIdx.x & 31;
  const int hl = lane >> 4;
  const int n = lane & 15;
  const unsigned short* p = smem + (col0 + n) * stride + k0;
  FragU f;
  f.u[0] = *(const uint4*)(p + 8 * hl);
  f.u[1] = *(const uint4*)(p + 16 + 8 * hl);
  return f.v;
}

__device__ inline v8f wmma_bf16(v16bf a, v16bf b, v8f c) {
  return __builtin_amdgcn_wmma_f32_16x16x32_bf16(false, a, false, b, (short)0, c, false, false);
}

// ---------------------------------------------------------------------------
// float -> bf16 conversion
// ---------------------------------------------------------------------------
__global__ __launch_bounds__(256) void f2bf_kernel(const float* __restrict__ in,
                                                   unsigned short* __restrict__ out, int n) {
  int i = blockIdx.x * 256 + threadIdx.x;
  if (i < n) out[i] = f32_to_bf16(in[i]);
}

// ---------------------------------------------------------------------------
// bf16 GEMM: C[M,N] = A[M,K] @ B[K,N] + bias.
// Block tile 128x128, K-step 32, 8 waves; each wave owns a 32x64 region
// (2 A-frags x 4 B-frags -> 8 WMMA per wave per K-step).
// mode 0: scatter into per-head Q/K/V bf16 buffers [bh, L, dk]   (N = 3072)
// mode 1: plain f32 row-major output                              (N = ldo)
// ---------------------------------------------------------------------------
__global__ __launch_bounds__(256) void gemm_bf16_kernel(
    const unsigned short* __restrict__ A, int lda,
    const unsigned short* __restrict__ B, int ldb,
    const float* __restrict__ bias, int K, int mode,
    float* __restrict__ outPlain, int ldo,
    unsigned short* __restrict__ Qd, unsigned short* __restrict__ Kd,
    unsigned short* __restrict__ Vd) {
  __shared__ __align__(16) unsigned short As[128 * 40];  // [row][k]  stride 40
  __shared__ __align__(16) unsigned short Bs[128 * 40];  // [n][k]    stride 40 (transposed)

  const int t = threadIdx.x;
  const int lane = t & 31;
  const int w = t >> 5;
  const int hl = lane >> 4;
  const int ln = lane & 15;
  const int wm = w >> 1;  // 0..3 -> 32-row strip
  const int wn = w & 1;   // 0..1 -> 64-col strip
  const int m0 = blockIdx.y * 128;
  const int n0 = blockIdx.x * 128;

  v8f acc[2][4];
#pragma unroll
  for (int im = 0; im < 2; ++im)
#pragma unroll
    for (int in = 0; in < 4; ++in) acc[im][in] = zero_v8f();

  const int ar = t >> 1, aseg = t & 1;  // 128 rows, 16 elems/thread
  const int br = t >> 3, bseg = t & 7;  // 32 K-rows, 16 elems/thread

  for (int kk = 0; kk < K; kk += 32) {
    {
      const uint4* g = (const uint4*)(A + (size_t)(m0 + ar) * lda + kk + aseg * 16);
      uint4* s = (uint4*)(&As[ar * 40 + aseg * 16]);
      s[0] = g[0];
      s[1] = g[1];
    }
    {
      const uint4* g = (const uint4*)(B + (size_t)(kk + br) * ldb + n0 + bseg * 16);
      uint4 d0 = g[0];
      uint4 d1 = g[1];
      const unsigned short* e0 = (const unsigned short*)&d0;
      const unsigned short* e1 = (const unsigned short*)&d1;
#pragma unroll
      for (int q = 0; q < 8; ++q) {
        Bs[(bseg * 16 + q) * 40 + br] = e0[q];
        Bs[(bseg * 16 + 8 + q) * 40 + br] = e1[q];
      }
    }
    if (kk + 32 < K) {  // prefetch next K-step (global_prefetch_b8)
      __builtin_prefetch(A + (size_t)(m0 + ar) * lda + kk + 32 + aseg * 16, 0, 1);
      __builtin_prefetch(B + (size_t)(kk + 32 + br) * ldb + n0 + bseg * 16, 0, 1);
    }
    __syncthreads();

    v16bf af[2], bfv[4];
#pragma unroll
    for (int im = 0; im < 2; ++im) af[im] = load_a_frag(As, wm * 32 + im * 16, 0, 40);
#pragma unroll
    for (int in = 0; in < 4; ++in) bfv[in] = load_bt_frag(Bs, wn * 64 + in * 16, 0, 40);
#pragma unroll
    for (int im = 0; im < 2; ++im)
#pragma unroll
      for (int in = 0; in < 4; ++in) acc[im][in] = wmma_bf16(af[im], bfv[in], acc[im][in]);
    __syncthreads();
  }

#pragma unroll
  for (int im = 0; im < 2; ++im)
#pragma unroll
    for (int in = 0; in < 4; ++in) {
      const int mBase = m0 + wm * 32 + im * 16 + hl * 8;
      const int nn = n0 + wn * 64 + in * 16 + ln;
      const float bv = bias[nn];
      if (mode == 0) {
        const int p = nn >> 10;        // 0=Q 1=K 2=V
        const int h = (nn >> 6) & 15;  // head
        const int dkk = nn & 63;       // within-head index
        unsigned short* dst = (p == 0) ? Qd : ((p == 1) ? Kd : Vd);
#pragma unroll
        for (int v = 0; v < 8; ++v) {
          const int m = mBase + v;
          const int b = m >> 11;
          const int l = m & 2047;
          dst[(((size_t)(b * 16 + h)) * 2048 + l) * 64 + dkk] = f32_to_bf16(acc[im][in][v] + bv);
        }
      } else {
#pragma unroll
        for (int v = 0; v < 8; ++v)
          outPlain[(size_t)(mBase + v) * ldo + nn] = acc[im][in][v] + bv;
      }
    }
}

// ---------------------------------------------------------------------------
// Per-row squared norms: sqq = sum(Qm^2+Qs^2), sqk = sum(Km^2+Ks^2)  [bh*L]
// ---------------------------------------------------------------------------
__global__ __launch_bounds__(256) void sumsq_kernel(
    const unsigned short* __restrict__ Qm, const unsigned short* __restrict__ Qs,
    const unsigned short* __restrict__ Km, const unsigned short* __restrict__ Ks,
    float* __restrict__ sqq, float* __restrict__ sqk) {
  const int idx = blockIdx.x * 256 + threadIdx.x;  // 65536 = bh*2048 + l
  const size_t base = (size_t)idx * 64;
  float aq = 0.f, ak = 0.f;
  for (int i = 0; i < 64; ++i) {
    float qm = bf16_to_f32(Qm[base + i]);
    float qs = bf16_to_f32(Qs[base + i]);
    float km = bf16_to_f32(Km[base + i]);
    float ks = bf16_to_f32(Ks[base + i]);
    aq += qm * qm + qs * qs;
    ak += km * km + ks * ks;
  }
  sqq[idx] = aq;
  sqk[idx] = ak;
}

// ---------------------------------------------------------------------------
// Fused Wasserstein attention, flash-style.
// Block = one (bh, 64-row Q tile). Streams 64-col K/V tiles.
// sim = exp(-W) <= 1  =>  exp(sim) <= e  => one-pass softmax, no max tracking.
// ---------------------------------------------------------------------------
__global__ __launch_bounds__(256) void wass_attn_kernel(
    const unsigned short* __restrict__ Qm, const unsigned short* __restrict__ Km,
    const unsigned short* __restrict__ Vm, const unsigned short* __restrict__ Qs,
    const unsigned short* __restrict__ Ks, const unsigned short* __restrict__ Vs,
    const float* __restrict__ sqq, const float* __restrict__ sqk,
    unsigned short* __restrict__ Omu, unsigned short* __restrict__ Osig) {
  __shared__ __align__(16) unsigned short Qms[64 * 72];
  __shared__ __align__(16) unsigned short Qss[64 * 72];
  __shared__ __align__(16) unsigned short Kms[64 * 72];
  __shared__ __align__(16) unsigned short Kss[64 * 72];
  __shared__ __align__(16) unsigned short Vmt[64 * 72];  // transposed [dk][j]
  __shared__ __align__(16) unsigned short Vst[64 * 72];  // transposed [dk][j]
  __shared__ __align__(16) unsigned short Ps[64 * 72];   // attention numerators, bf16
  __shared__ float sqqS[64];
  __shared__ float sqkS[64];
  __shared__ float rowS[64];

  const int t = threadIdx.x;
  const int lane = t & 31;
  const int w = t >> 5;
  const int hl = lane >> 4;
  const int ln = lane & 15;
  const int wm = w >> 1;
  const int wn = w & 1;
  const int bh = blockIdx.y;       // b*16 + h
  const int m0 = blockIdx.x * 64;  // Q row tile
  const size_t headBase = (size_t)bh * 2048 * 64;

  {  // stage Q tiles once
    const int r = t >> 2, seg = t & 3;
    const uint4* gm = (const uint4*)(Qm + headBase + (size_t)(m0 + r) * 64 + seg * 16);
    const uint4* gs = (const uint4*)(Qs + headBase + (size_t)(m0 + r) * 64 + seg * 16);
    uint4* sm = (uint4*)(&Qms[r * 72 + seg * 16]);
    uint4* ss = (uint4*)(&Qss[r * 72 + seg * 16]);
    sm[0] = gm[0]; sm[1] = gm[1];
    ss[0] = gs[0]; ss[1] = gs[1];
  }
  if (t < 64) {
    sqqS[t] = sqq[(size_t)bh * 2048 + m0 + t];
    rowS[t] = 0.f;
  }

  v8f aOm[2][2], aOs[2][2];
#pragma unroll
  for (int im = 0; im < 2; ++im)
#pragma unroll
    for (int in = 0; in < 2; ++in) {
      aOm[im][in] = zero_v8f();
      aOs[im][in] = zero_v8f();
    }

  for (int j0 = 0; j0 < 2048; j0 += 64) {
    {  // stage K tiles [j][d]
      const int r = t >> 2, seg = t & 3;
      const uint4* gm = (const uint4*)(Km + headBase + (size_t)(j0 + r) * 64 + seg * 16);
      const uint4* gs = (const uint4*)(Ks + headBase + (size_t)(j0 + r) * 64 + seg * 16);
      uint4* sm = (uint4*)(&Kms[r * 72 + seg * 16]);
      uint4* ss = (uint4*)(&Kss[r * 72 + seg * 16]);
      sm[0] = gm[0]; sm[1] = gm[1];
      ss[0] = gs[0]; ss[1] = gs[1];
    }
    {  // stage V tiles transposed [dk][j]
      const int seg = t & 7;
      for (int r = t >> 3; r < 64; r += 32) {
        uint4 dm = *(const uint4*)(Vm + headBase + (size_t)(j0 + r) * 64 + seg * 8);
        uint4 dsv = *(const uint4*)(Vs + headBase + (size_t)(j0 + r) * 64 + seg * 8);
        const unsigned short* em = (const unsigned short*)&dm;
        const unsigned short* es = (const unsigned short*)&dsv;
#pragma unroll
        for (int q = 0; q < 8; ++q) {
          Vmt[(seg * 8 + q) * 72 + r] = em[q];
          Vst[(seg * 8 + q) * 72 + r] = es[q];
        }
      }
    }
    if (t < 64) sqkS[t] = sqk[(size_t)bh * 2048 + j0 + t];
    if (j0 + 64 < 2048) {  // prefetch next K/V tiles (global_prefetch_b8)
      const size_t nb = headBase + (size_t)(j0 + 64 + (t >> 2)) * 64;
      __builtin_prefetch(Km + nb, 0, 1);
      __builtin_prefetch(Ks + nb, 0, 1);
      __builtin_prefetch(Vm + nb, 0, 1);
      __builtin_prefetch(Vs + nb, 0, 1);
    }
    __syncthreads();

    // S = Qm*Km^T + Qs*Ks^T  (64x64 tile, K-dim = dk = 64)
    v8f accS[2][2];
#pragma unroll
    for (int im = 0; im < 2; ++im)
#pragma unroll
      for (int in = 0; in < 2; ++in) accS[im][in] = zero_v8f();

#pragma unroll
    for (int k0 = 0; k0 < 64; k0 += 32) {
      v16bf am[2], as2[2], bm[2], bs2[2];
#pragma unroll
      for (int im = 0; im < 2; ++im) {
        am[im] = load_a_frag(Qms, wm * 32 + im * 16, k0, 72);
        as2[im] = load_a_frag(Qss, wm * 32 + im * 16, k0, 72);
      }
#pragma unroll
      for (int in = 0; in < 2; ++in) {
        bm[in] = load_bt_frag(Kms, wn * 32 + in * 16, k0, 72);
        bs2[in] = load_bt_frag(Kss, wn * 32 + in * 16, k0, 72);
      }
#pragma unroll
      for (int im = 0; im < 2; ++im)
#pragma unroll
        for (int in = 0; in < 2; ++in) {
          accS[im][in] = wmma_bf16(am[im], bm[in], accS[im][in]);
          accS[im][in] = wmma_bf16(as2[im], bs2[in], accS[im][in]);
        }
    }

    // W2 -> W -> sim -> p = exp(sim); store P tile (bf16)
#pragma unroll
    for (int im = 0; im < 2; ++im)
#pragma unroll
      for (int in = 0; in < 2; ++in) {
        const int ml = wm * 32 + im * 16 + hl * 8;
        const int nl = wn * 32 + in * 16 + ln;
        const float skj = sqkS[nl];
#pragma unroll
        for (int v = 0; v < 8; ++v) {
          const float s = accS[im][in][v];
          const float w2 = sqqS[ml + v] + skj - 2.f * s;
          const float wd = sqrtf(fmaxf(w2, 0.f));
          const float pp = __expf(__expf(-wd));  // exp(similarity), sim in (0,1]
          Ps[(ml + v) * 72 + nl] = f32_to_bf16(pp);
        }
      }
    __syncthreads();

    // Deterministic row sums of P (use the same bf16-rounded values WMMA consumes)
    if (t < 64) {
      float s = 0.f;
      for (int j = 0; j < 64; ++j) s += bf16_to_f32(Ps[t * 72 + j]);
      rowS[t] += s;
    }

    // O += P @ V (mu and sigma)
#pragma unroll
    for (int k0 = 0; k0 < 64; k0 += 32) {
      v16bf ap[2], bvm[2], bvs[2];
#pragma unroll
      for (int im = 0; im < 2; ++im) ap[im] = load_a_frag(Ps, wm * 32 + im * 16, k0, 72);
#pragma unroll
      for (int in = 0; in < 2; ++in) {
        bvm[in] = load_bt_frag(Vmt, wn * 32 + in * 16, k0, 72);
        bvs[in] = load_bt_frag(Vst, wn * 32 + in * 16, k0, 72);
      }
#pragma unroll
      for (int im = 0; im < 2; ++im)
#pragma unroll
        for (int in = 0; in < 2; ++in) {
          aOm[im][in] = wmma_bf16(ap[im], bvm[in], aOm[im][in]);
          aOs[im][in] = wmma_bf16(ap[im], bvs[in], aOs[im][in]);
        }
    }
    __syncthreads();
  }

  // normalize and write merged-head outputs [b, l, h*64+dk] as bf16
  const int b = bh >> 4;
  const int h = bh & 15;
#pragma unroll
  for (int im = 0; im < 2; ++im)
#pragma unroll
    for (int in = 0; in < 2; ++in) {
      const int ml = wm * 32 + im * 16 + hl * 8;
      const int nl = wn * 32 + in * 16 + ln;
#pragma unroll
      for (int v = 0; v < 8; ++v) {
        const float inv = 1.f / rowS[ml + v];
        const size_t oi = ((size_t)(b * 2048 + m0 + ml + v)) * 1024 + h * 64 + nl;
        Omu[oi] = f32_to_bf16(aOm[im][in][v] * inv);
        Osig[oi] = f32_to_bf16(aOs[im][in][v] * inv);
      }
    }
}

// ---------------------------------------------------------------------------
extern "C" void kernel_launch(void* const* d_in, const int* in_sizes, int n_in,
                              void* d_out, int out_size, void* d_ws, size_t ws_size,
                              hipStream_t stream) {
  (void)in_sizes; (void)n_in; (void)out_size; (void)ws_size;
  const float* mu = (const float*)d_in[0];
  const float* sigma = (const float*)d_in[1];
  const float* Wqm = (const float*)d_in[2];
  const float* bqm = (const float*)d_in[3];
  const float* Wqs = (const float*)d_in[4];
  const float* bqs = (const float*)d_in[5];
  const float* Wo = (const float*)d_in[6];
  const float* bo = (const float*)d_in[7];
  float* out = (float*)d_out;

  char* ws = (char*)d_ws;
  size_t off = 0;
  auto take = [&](size_t bytes) -> void* {
    void* p = ws + off;
    off += (bytes + 255) & ~(size_t)255;
    return p;
  };

  const size_t NX = 4096ull * 1024;         // mu/sigma/O elements
  const size_t NWQ = 1024ull * 3072;        // qkv weight elements
  const size_t NWO = 1024ull * 1024;        // out weight elements
  const size_t NH = 2ull * 16 * 2048 * 64;  // per-head buffer elements

  unsigned short* muB = (unsigned short*)take(NX * 2);
  unsigned short* sgB = (unsigned short*)take(NX * 2);
  unsigned short* WqmB = (unsigned short*)take(NWQ * 2);
  unsigned short* WqsB = (unsigned short*)take(NWQ * 2);
  unsigned short* WoB = (unsigned short*)take(NWO * 2);
  unsigned short* QmB = (unsigned short*)take(NH * 2);
  unsigned short* KmB = (unsigned short*)take(NH * 2);
  unsigned short* VmB = (unsigned short*)take(NH * 2);
  unsigned short* QsB = (unsigned short*)take(NH * 2);
  unsigned short* KsB = (unsigned short*)take(NH * 2);
  unsigned short* VsB = (unsigned short*)take(NH * 2);
  float* sqqB = (float*)take(65536ull * 4);
  float* sqkB = (float*)take(65536ull * 4);
  unsigned short* OmuB = (unsigned short*)take(NX * 2);
  unsigned short* OsgB = (unsigned short*)take(NX * 2);

  f2bf_kernel<<<(int)((NX + 255) / 256), 256, 0, stream>>>(mu, muB, (int)NX);
  f2bf_kernel<<<(int)((NX + 255) / 256), 256, 0, stream>>>(sigma, sgB, (int)NX);
  f2bf_kernel<<<(int)((NWQ + 255) / 256), 256, 0, stream>>>(Wqm, WqmB, (int)NWQ);
  f2bf_kernel<<<(int)((NWQ + 255) / 256), 256, 0, stream>>>(Wqs, WqsB, (int)NWQ);
  f2bf_kernel<<<(int)((NWO + 255) / 256), 256, 0, stream>>>(Wo, WoB, (int)NWO);

  dim3 gq(24, 32);  // N=3072/128, M=4096/128
  gemm_bf16_kernel<<<gq, 256, 0, stream>>>(muB, 1024, WqmB, 3072, bqm, 1024, 0,
                                           nullptr, 0, QmB, KmB, VmB);
  gemm_bf16_kernel<<<gq, 256, 0, stream>>>(sgB, 1024, WqsB, 3072, bqs, 1024, 0,
                                           nullptr, 0, QsB, KsB, VsB);

  sumsq_kernel<<<256, 256, 0, stream>>>(QmB, QsB, KmB, KsB, sqqB, sqkB);

  dim3 ga(32, 32);  // L/64 row tiles, bs*H
  wass_attn_kernel<<<ga, 256, 0, stream>>>(QmB, KmB, VmB, QsB, KsB, VsB, sqqB, sqkB,
                                           OmuB, OsgB);

  dim3 go(8, 32);  // N=1024/128, M=4096/128
  gemm_bf16_kernel<<<go, 256, 0, stream>>>(OmuB, 1024, WoB, 1024, bo, 1024, 1,
                                           out, 1024, nullptr, nullptr, nullptr);
  gemm_bf16_kernel<<<go, 256, 0, stream>>>(OsgB, 1024, WoB, 1024, bo, 1024, 1,
                                           out + NX, 1024, nullptr, nullptr, nullptr);
}
